// SelfAttention_48885317763470
// MI455X (gfx1250) — compile-verified, matrix operand order checked by reading, not run on hardware
//
#include <hip/hip_runtime.h>
#include <math.h>

// ---------------------------------------------------------------------------
// Problem constants (B=2, L=2048, DIM=2048, HEADS=16, HD=128)
// ---------------------------------------------------------------------------
#define B_      2
#define L_      2048
#define DIM_    2048
#define HEADS_  16
#define HD_     128
#define BL_     (B_ * L_)        // 4096 rows
#define BH_     (B_ * HEADS_)    // 32 (batch*heads)

typedef __bf16 bf16_t;
typedef __attribute__((ext_vector_type(16))) __bf16 v16bf;
typedef __attribute__((ext_vector_type(8)))  __bf16 v8bf;
typedef __attribute__((ext_vector_type(8)))  float  v8f;

// round-to-nearest-even f32 -> bf16 (bit-level)
static __device__ inline bf16_t to_bf16(float f) {
    unsigned u = __float_as_uint(f);
    unsigned r = (u + 0x7FFFu + ((u >> 16) & 1u)) >> 16;
    unsigned short s = (unsigned short)r;
    bf16_t b;
    __builtin_memcpy(&b, &s, 2);
    return b;
}

// Load one fragment: 16 bf16 per lane as two contiguous 16B chunks.
// Matches CDNA5 16-bit A/B layout (ISA 7.12.2): lane holds k in
// [kg*8, kg*8+8) U [16+kg*8, 16+kg*8+8).
static __device__ inline v16bf load_frag16(const bf16_t* p, int kg) {
    v8bf lo = *(const v8bf*)(p + kg * 8);
    v8bf hi = *(const v8bf*)(p + 16 + kg * 8);
    v16bf r;
#pragma unroll
    for (int i = 0; i < 8; ++i) { r[i] = lo[i]; r[i + 8] = hi[i]; }
    return r;
}

// ---------------------------------------------------------------------------
// Kernel 0: fp32 -> bf16 cast
// ---------------------------------------------------------------------------
__global__ __launch_bounds__(256) void cast_f32_to_bf16(
    const float* __restrict__ in, bf16_t* __restrict__ out, int n) {
    int i = blockIdx.x * 256 + threadIdx.x;
    if (i < n) out[i] = to_bf16(in[i]);
}

// ---------------------------------------------------------------------------
// Kernels 1 & 4: C[M,N] = A[M,K] * B[N,K]^T  (both bf16 k-contiguous), fp32 out.
// One wave computes a 32x64 tile: 2 A-frags x 4 B-frags -> 8 WMMAs per k-step
// (0.75 fragment loads per WMMA). ~130 VGPRs/lane.
// ---------------------------------------------------------------------------
template <bool HAS_BIAS>
__global__ __launch_bounds__(256) void gemm_bf16_nt(
    const bf16_t* __restrict__ A, const bf16_t* __restrict__ Bm,
    const float* __restrict__ bias, float* __restrict__ C,
    int M, int N, int K) {
    int wave   = blockIdx.x * 8 + (threadIdx.x >> 5);
    int tilesN = N >> 6;                    // 64-wide n tiles
    int tiles  = (M >> 5) * tilesN;
    if (wave >= tiles) return;
    int tm = (wave / tilesN) * 32;
    int tn = (wave % tilesN) * 64;

    int lane = threadIdx.x & 31;
    int rc = lane & 15;                     // row (A) / col (B) within subtile
    int kg = (lane >> 4) & 1;               // k-subgroup select

    v8f acc[2][4];
#pragma unroll
    for (int i = 0; i < 2; ++i)
#pragma unroll
        for (int j = 0; j < 4; ++j) acc[i][j] = (v8f){};

    const bf16_t* ap[2];
    const bf16_t* bp[4];
#pragma unroll
    for (int i = 0; i < 2; ++i) ap[i] = A + (size_t)(tm + i * 16 + rc) * K;
#pragma unroll
    for (int j = 0; j < 4; ++j) bp[j] = Bm + (size_t)(tn + j * 16 + rc) * K;

    for (int k0 = 0; k0 < K; k0 += 32) {
        if (k0 + 64 < K) {                  // prefetch two k-steps ahead
            __builtin_prefetch(ap[0] + k0 + 64, 0, 0);
            __builtin_prefetch(bp[0] + k0 + 64, 0, 0);
        }
        v16bf af[2], bf[4];
#pragma unroll
        for (int i = 0; i < 2; ++i) af[i] = load_frag16(ap[i] + k0, kg);
#pragma unroll
        for (int j = 0; j < 4; ++j) bf[j] = load_frag16(bp[j] + k0, kg);
#pragma unroll
        for (int i = 0; i < 2; ++i)
#pragma unroll
            for (int j = 0; j < 4; ++j)
                acc[i][j] = __builtin_amdgcn_wmma_f32_16x16x32_bf16(
                    false, af[i], false, bf[j], (short)0, acc[i][j], false, false);
    }

    // C/D layout: VGPR r, lane -> (m = r + (lane<16?0:8), n = lane&15)
    int mo = (lane < 16) ? 0 : 8;
    float bia[4] = {0.f, 0.f, 0.f, 0.f};
    if (HAS_BIAS) {
#pragma unroll
        for (int j = 0; j < 4; ++j) bia[j] = bias[tn + j * 16 + rc];
    }
#pragma unroll
    for (int i = 0; i < 2; ++i)
#pragma unroll
        for (int r = 0; r < 8; ++r) {
            size_t rowb = (size_t)(tm + i * 16 + mo + r) * N;
#pragma unroll
            for (int j = 0; j < 4; ++j)
                C[rowb + tn + j * 16 + rc] = acc[i][j][r] + bia[j];
        }
}

// ---------------------------------------------------------------------------
// Kernel 2: RMSNorm (per head over HD) + RoPE + repack to bf16.
//   q,k -> [BH, L, HD]   (d-contiguous: serves as A rows and B cols of Q@K^T)
//   v   -> [BH, HD, L]   (transposed: WMMA B-matrix for P@V, contiguous in key)
// ---------------------------------------------------------------------------
__global__ __launch_bounds__(128) void norm_rope_pack(
    const float* __restrict__ qkv, const float* __restrict__ pe,
    const float* __restrict__ q_scale, const float* __restrict__ k_scale,
    bf16_t* __restrict__ qo, bf16_t* __restrict__ ko, bf16_t* __restrict__ vT) {
    int bl = blockIdx.x;            // 0..B*L-1
    int h  = blockIdx.y;            // 0..HEADS-1
    int b  = bl / L_;
    int l  = bl % L_;
    int d  = threadIdx.x;           // 0..127

    size_t base = (size_t)bl * (3 * DIM_);
    float q = qkv[base + (size_t)h * HD_ + d];
    float k = qkv[base + DIM_ + (size_t)h * HD_ + d];
    float v = qkv[base + 2 * DIM_ + (size_t)h * HD_ + d];

    __shared__ float red[2][4];
    __shared__ float qn[HD_], kn[HD_];

    float sq = q * q, sk = k * k;
#pragma unroll
    for (int m = 1; m < 32; m <<= 1) {
        sq += __shfl_xor(sq, m, 32);
        sk += __shfl_xor(sk, m, 32);
    }
    int w = threadIdx.x >> 5;
    if ((threadIdx.x & 31) == 0) { red[0][w] = sq; red[1][w] = sk; }
    __syncthreads();
    sq = red[0][0] + red[0][1] + red[0][2] + red[0][3];
    sk = red[1][0] + red[1][1] + red[1][2] + red[1][3];
    float rq = rsqrtf(sq * (1.0f / HD_) + 1e-8f);
    float rk = rsqrtf(sk * (1.0f / HD_) + 1e-8f);
    qn[d] = q * rq * q_scale[d];
    kn[d] = k * rk * k_scale[d];
    __syncthreads();

    // RoPE: out[2i+j] = pe[l,i,j,0]*t[2i] + pe[l,i,j,1]*t[2i+1]
    int i  = d >> 1, jj = d & 1;
    size_t peb = (((size_t)l * (HD_ / 2) + i) * 2 + jj) * 2;
    float e0 = pe[peb + 0], e1 = pe[peb + 1];
    float qr = e0 * qn[2 * i] + e1 * qn[2 * i + 1];
    float kr = e0 * kn[2 * i] + e1 * kn[2 * i + 1];

    size_t bh = (size_t)b * HEADS_ + h;
    qo[(bh * L_ + l) * HD_ + d] = to_bf16(qr);
    ko[(bh * L_ + l) * HD_ + d] = to_bf16(kr);
    vT[(bh * HD_ + d) * L_ + l] = to_bf16(v);
}

// ---------------------------------------------------------------------------
// Kernel 3: flash attention. Block = 128 threads = 4 waves; each wave owns a
// 16-row Q tile (whole HD resident in 4 A-frags). Per 32-key step the block
// cooperatively stages the shared 8KB K chunk + 8KB V^T chunk into LDS (4x
// less global/L2 traffic than per-wave loads), then each wave:
//   8 WMMAs (S = Q K^T, B-frags from LDS) -> online softmax (shfl reductions
//   inside the 16-lane C-matrix halves) -> P through per-wave LDS staging
//   (C-layout -> A-layout; LDS ops of one wave are in-order per the ISA)
//   -> 8 WMMAs (O += P V, B-frags from LDS).
// ---------------------------------------------------------------------------
__global__ __launch_bounds__(128) void flash_attn(
    const bf16_t* __restrict__ Q, const bf16_t* __restrict__ Kmat,
    const bf16_t* __restrict__ Vt, bf16_t* __restrict__ O) {
    int bh   = blockIdx.y;
    int b    = bh / HEADS_;
    int h    = bh % HEADS_;
    int wid  = threadIdx.x >> 5;
    int lane = threadIdx.x & 31;
    int rc   = lane & 15;
    int kg   = (lane >> 4) & 1;
    int q0   = blockIdx.x * 64 + wid * 16;

    const bf16_t* qb = Q    + ((size_t)bh * L_ + q0) * HD_;
    const bf16_t* kb = Kmat + (size_t)bh * L_ * HD_;
    const bf16_t* vb = Vt   + (size_t)bh * HD_ * L_;

    // Whole Q tile resident: 16 rows x HD=128 -> 4 A-frags
    v16bf qa[4];
#pragma unroll
    for (int c = 0; c < 4; ++c)
        qa[c] = load_frag16(qb + (size_t)rc * HD_ + c * 32, kg);

    v8f acc[8];
#pragma unroll
    for (int t = 0; t < 8; ++t) acc[t] = (v8f){};
    float mrow[8], lrow[8];
#pragma unroll
    for (int r = 0; r < 8; ++r) { mrow[r] = -INFINITY; lrow[r] = 0.f; }

    __shared__ __align__(16) bf16_t ks[32 * HD_];      // K chunk  [key][d]  8KB
    __shared__ __align__(16) bf16_t vs[HD_][32];       // V^T chunk [d][key] 8KB
    __shared__ __align__(16) bf16_t pst[4][16][32];    // per-wave P staging 4KB
    const float scale = 0.08838834764831845f;          // 1/sqrt(128)

    for (int kv = 0; kv < L_; kv += 32) {
        __syncthreads();   // previous iteration's LDS reads complete
        // ---- cooperative staging: K chunk is 8KB contiguous; V^T rows strided
        {
            const bf16_t* ksrc = kb + (size_t)kv * HD_;
#pragma unroll
            for (int rep = 0; rep < 4; ++rep) {
                int idx = rep * 128 + threadIdx.x;            // 16B chunk id
                *(v8bf*)&ks[idx * 8] = *(const v8bf*)(ksrc + idx * 8);
            }
#pragma unroll
            for (int rep = 0; rep < 4; ++rep) {
                int idx = rep * 128 + threadIdx.x;            // 0..511
                int d = idx >> 2, c = idx & 3;                // 4x16B per row
                *(v8bf*)&vs[d][c * 8] =
                    *(const v8bf*)(vb + (size_t)d * L_ + kv + c * 8);
            }
        }
        __syncthreads();   // staging visible to all waves

        // ---- S = Q K^T for 32 keys (two 16-wide N tiles), B-frags from LDS
        v8f s0 = {}, s1 = {};
#pragma unroll
        for (int c = 0; c < 4; ++c) {
            v16bf bk0 = load_frag16(&ks[(size_t)rc * HD_ + c * 32], kg);
            v16bf bk1 = load_frag16(&ks[(size_t)(16 + rc) * HD_ + c * 32], kg);
            s0 = __builtin_amdgcn_wmma_f32_16x16x32_bf16(false, qa[c], false, bk0, (short)0, s0, false, false);
            s1 = __builtin_amdgcn_wmma_f32_16x16x32_bf16(false, qa[c], false, bk1, (short)0, s1, false, false);
        }

        // ---- online softmax; rows live in (VGPR r, lane-half), cols across lanes
        float alpha[8];
#pragma unroll
        for (int r = 0; r < 8; ++r) {
            float a0 = s0[r] * scale, a1 = s1[r] * scale;
            float mx = fmaxf(a0, a1);
#pragma unroll
            for (int m = 1; m < 16; m <<= 1) mx = fmaxf(mx, __shfl_xor(mx, m, 32));
            float mn = fmaxf(mrow[r], mx);
            alpha[r] = __expf(mrow[r] - mn);
            mrow[r]  = mn;
            float p0 = __expf(a0 - mn), p1 = __expf(a1 - mn);
            s0[r] = p0; s1[r] = p1;
            float ps = p0 + p1;
#pragma unroll
            for (int m = 1; m < 16; m <<= 1) ps += __shfl_xor(ps, m, 32);
            lrow[r] = lrow[r] * alpha[r] + ps;
        }
#pragma unroll
        for (int t = 0; t < 8; ++t)
#pragma unroll
            for (int r = 0; r < 8; ++r) acc[t][r] *= alpha[r];

        // ---- P: C-layout -> per-wave LDS -> A-layout (wave-private staging;
        //      LDS ops of one wave execute in order, so no block barrier needed)
        int mo = (lane < 16) ? 0 : 8;
#pragma unroll
        for (int r = 0; r < 8; ++r) {
            pst[wid][mo + r][rc]      = to_bf16(s0[r]);
            pst[wid][mo + r][16 + rc] = to_bf16(s1[r]);
        }
        __builtin_amdgcn_wave_barrier();   // scheduling fence only
        v16bf pa = load_frag16(&pst[wid][rc][0], kg);

        // ---- O += P V : staged V^T rows are B-columns (contiguous over keys)
#pragma unroll
        for (int t = 0; t < 8; ++t) {
            v16bf vf = load_frag16(&vs[t * 16 + rc][0], kg);
            acc[t] = __builtin_amdgcn_wmma_f32_16x16x32_bf16(false, pa, false, vf, (short)0, acc[t], false, false);
        }
    }

    // ---- epilogue: normalize and pack into [B, L, DIM] bf16 ----
    int mo = (lane < 16) ? 0 : 8;
#pragma unroll
    for (int t = 0; t < 8; ++t) {
#pragma unroll
        for (int r = 0; r < 8; ++r) {
            int m = q0 + mo + r;
            int d = t * 16 + rc;
            float o = acc[t][r] / lrow[r];
            O[((size_t)b * L_ + m) * DIM_ + (size_t)h * HD_ + d] = to_bf16(o);
        }
    }
}

// ---------------------------------------------------------------------------
// Host-side launcher
// ---------------------------------------------------------------------------
extern "C" void kernel_launch(void* const* d_in, const int* in_sizes, int n_in,
                              void* d_out, int out_size, void* d_ws, size_t ws_size,
                              hipStream_t stream) {
    (void)in_sizes; (void)n_in; (void)out_size; (void)ws_size;
    const float* x       = (const float*)d_in[0];
    const float* pe      = (const float*)d_in[1];
    const float* w_qkv   = (const float*)d_in[2];
    const float* q_scale = (const float*)d_in[3];
    const float* k_scale = (const float*)d_in[4];
    const float* w_proj  = (const float*)d_in[5];
    const float* b_proj  = (const float*)d_in[6];
    float* out = (float*)d_out;

    // workspace carve-up (sizes are multiples of 4 KB; base is 256B-aligned)
    char* ws = (char*)d_ws;
    bf16_t* x_bf     = (bf16_t*)ws; ws += (size_t)BL_ * DIM_ * 2;          // 16 MB
    bf16_t* wqkv_bf  = (bf16_t*)ws; ws += (size_t)3 * DIM_ * DIM_ * 2;     // 25 MB
    bf16_t* wproj_bf = (bf16_t*)ws; ws += (size_t)DIM_ * DIM_ * 2;         //  8 MB
    float*  qkv_f    = (float*)ws;  ws += (size_t)BL_ * 3 * DIM_ * 4;      // 100 MB
    bf16_t* q_bf     = (bf16_t*)ws; ws += (size_t)BH_ * L_ * HD_ * 2;      // 16 MB
    bf16_t* k_bf     = (bf16_t*)ws; ws += (size_t)BH_ * L_ * HD_ * 2;      // 16 MB
    bf16_t* vT_bf    = (bf16_t*)ws; ws += (size_t)BH_ * HD_ * L_ * 2;      // 16 MB
    bf16_t* o_bf     = (bf16_t*)ws; ws += (size_t)BL_ * DIM_ * 2;          // 16 MB

    // 0) cast operands to bf16
    {
        int n0 = BL_ * DIM_;            // x
        int n1 = 3 * DIM_ * DIM_;       // w_qkv
        int n2 = DIM_ * DIM_;           // w_proj
        cast_f32_to_bf16<<<(n0 + 255) / 256, 256, 0, stream>>>(x, x_bf, n0);
        cast_f32_to_bf16<<<(n1 + 255) / 256, 256, 0, stream>>>(w_qkv, wqkv_bf, n1);
        cast_f32_to_bf16<<<(n2 + 255) / 256, 256, 0, stream>>>(w_proj, wproj_bf, n2);
    }

    // 1) qkv = x @ w_qkv^T   (M=4096, N=6144, K=2048) -> fp32
    {
        int tiles = (BL_ / 32) * ((3 * DIM_) / 64);   // 12288 wave tiles
        gemm_bf16_nt<false><<<tiles / 8, 256, 0, stream>>>(
            x_bf, wqkv_bf, nullptr, qkv_f, BL_, 3 * DIM_, DIM_);
    }

    // 2) RMSNorm + RoPE + repack (q,k -> [BH,L,HD]; v -> [BH,HD,L])
    norm_rope_pack<<<dim3(BL_, HEADS_), 128, 0, stream>>>(
        qkv_f, pe, q_scale, k_scale, q_bf, k_bf, vT_bf);

    // 3) flash attention -> o_bf [B, L, DIM]
    flash_attn<<<dim3(L_ / 64, BH_), 128, 0, stream>>>(q_bf, k_bf, vT_bf, o_bf);

    // 4) out = o @ w_proj^T + b_proj   (M=4096, N=2048, K=2048) -> fp32
    {
        int tiles = (BL_ / 32) * (DIM_ / 64);         // 4096 wave tiles
        gemm_bf16_nt<true><<<tiles / 8, 256, 0, stream>>>(
            o_bf, wproj_bf, b_proj, out, BL_, DIM_, DIM_);
    }
}
